// UniformHeaderGraphAttentionLayer_61615600828790
// MI455X (gfx1250) — compile-verified, matrix operand order checked by reading, not run on hardware
//
#include <hip/hip_runtime.h>
#include <hip/hip_bf16.h>
#include <stdint.h>

// GAT layer for MI455X (gfx1250), fully fused, no online softmax.
// Shapes fixed by harness: N=8192 nodes, D=64 features.
//   scores s_ij = mask(adj, leakyrelu(wh1_i + wh2_j)); row softmax; (P @ h); relu.
// Row-max upper bound M_i = leakyrelu(wh1_i + max_j wh2_j) (leaky monotone) => fixed-shift
// softmax: p_ij = exp(s_ij - M_i), no running max/rescale, no per-tile reductions.
// HBM floor: adj (256 MB) read exactly once => ~11us @ 23.3 TB/s. h/bfrag (3MB) live in L2.

typedef __attribute__((ext_vector_type(16))) __bf16 v16bf;
typedef __attribute__((ext_vector_type(8)))  float  v8f;

#define GAT_N     8192
#define GAT_D     64
#define GAT_ALPHA 0.2f
#define LOG2E     1.4426950408889634f
#define EXP2_NEG  -1.0e38f       // exp2(-1e38) == 0 -> masked entries vanish
#define P_STRIDE  272            // bytes per P row: 128 bf16 + 8 bf16 pad (conflict-free b128)
#define P_BUF     (16 * P_STRIDE)

static __device__ __forceinline__ uint32_t f2bf_bits(float x) {
    return (uint32_t)__builtin_bit_cast(unsigned short, (__bf16)x);
}
static __device__ __forceinline__ float leaky(float e) {
    return fmaxf(e, GAT_ALPHA * e);   // branchless LeakyReLU: v_mul + v_max
}

// ---------------- Prologue 1: wh1[i] = h[i,:].a[:64], wh2[i] = h[i,:].a[64:] ----------------
__global__ void gat_wh_kernel(const float* __restrict__ h, const float* __restrict__ a,
                              float* __restrict__ wh1, float* __restrict__ wh2) {
    int i = blockIdx.x * blockDim.x + threadIdx.x;
    if (i >= GAT_N) return;
    const float4* hv = (const float4*)(h + (size_t)i * GAT_D);
    float s1 = 0.f, s2 = 0.f;
#pragma unroll
    for (int c = 0; c < 16; ++c) {
        float4 v = hv[c];
        s1 += v.x * a[4*c + 0]         + v.y * a[4*c + 1]
            + v.z * a[4*c + 2]         + v.w * a[4*c + 3];
        s2 += v.x * a[GAT_D + 4*c + 0] + v.y * a[GAT_D + 4*c + 1]
            + v.z * a[GAT_D + 4*c + 2] + v.w * a[GAT_D + 4*c + 3];
    }
    wh1[i] = s1;
    wh2[i] = s2;
}

// ---------------- Prologue 2: global max of wh2 (single workgroup) ----------------
__global__ void gat_w2max_kernel(const float* __restrict__ wh2, float* __restrict__ w2max) {
    __shared__ float red[256];
    float m = -__builtin_inff();
    for (int i = threadIdx.x; i < GAT_N; i += 256) m = fmaxf(m, wh2[i]);
    red[threadIdx.x] = m;
    __syncthreads();
#pragma unroll
    for (int s = 128; s > 0; s >>= 1) {
        if ((int)threadIdx.x < s) red[threadIdx.x] = fmaxf(red[threadIdx.x], red[threadIdx.x + s]);
        __syncthreads();
    }
    if (threadIdx.x == 0) *w2max = red[0];
}

// ---------------- Prologue 3: pre-swizzle h (f32->bf16) into WMMA B-fragment layout ----------
// Per 32-wide K-tile kt, per 16-col slice w: lane {g=lane>>4, n=lane&15}, VGPR v packs bf16
// pair (K0,K0+1), K0 = 2v + (v>=4 ? 8 : 0) + 8g. Lane fetches its 8 dwords as two b128 loads.
__global__ void gat_bfrag_kernel(const float* __restrict__ h, uint32_t* __restrict__ bfrag) {
    int tid = blockIdx.x * blockDim.x + threadIdx.x;   // one per (kt, w, lane)
    if (tid >= (GAT_N / 32) * 4 * 32) return;
    int l = tid & 31;
    int w = (tid >> 5) & 3;
    int kt = tid >> 7;
    int g = l >> 4;
    int n = 16 * w + (l & 15);
    int j0 = 32 * kt;
    uint32_t d[8];
#pragma unroll
    for (int v = 0; v < 8; ++v) {
        int K0 = 2 * v + ((v >= 4) ? 8 : 0) + 8 * g;
        float x0 = h[(size_t)(j0 + K0    ) * GAT_D + n];
        float x1 = h[(size_t)(j0 + K0 + 1) * GAT_D + n];
        d[v] = f2bf_bits(x0) | (f2bf_bits(x1) << 16);
    }
    uint4* out = (uint4*)(bfrag + (size_t)tid * 8);
    out[0] = make_uint4(d[0], d[1], d[2], d[3]);
    out[1] = make_uint4(d[4], d[5], d[6], d[7]);
}

// ---------------- Main fused kernel: 16 rows/block, 4 waves, K-tile = 128 ----------------
__global__ void __launch_bounds__(128) gat_main_kernel(
        const float* __restrict__ wh1, const float* __restrict__ wh2,
        const float* __restrict__ w2max,
        const int* __restrict__ adj, const uint32_t* __restrict__ bfrag,
        float* __restrict__ out) {
    __shared__ __align__(16) unsigned char Pl[2][P_BUF];  // double-buffered bf16 P tile
    __shared__ float Ll[16];                              // softmax denominators

    const int lane = threadIdx.x & 31;
    const int wv   = threadIdx.x >> 5;   // owns rows 4wv..4wv+3 and output cols 16wv..16wv+15
    const int lg   = lane >> 4;
    const int ln   = lane & 15;
    const int i0   = blockIdx.x * 16;

    const float w2m = *w2max;
    float wh1r[4], nMl[4], lr[4];
#pragma unroll
    for (int r = 0; r < 4; ++r) {
        wh1r[r] = wh1[i0 + 4 * wv + r];
        nMl[r]  = -leaky(wh1r[r] + w2m) * LOG2E;  // -M_i * log2e, folded into one v_fma
        lr[r]   = 0.f;
    }

    // Induction pointers: all in-loop accesses use compile-time immediate offsets.
    const int*   aptr  = adj + (size_t)(i0 + 4 * wv) * GAT_N + 4 * lane;  // += 128/iter
    const float* w2ptr = wh2 + 4 * lane;                                  // += 128/iter
    const char*  bptr  = (const char*)bfrag + (wv * 1024 + lane * 32);    // += 16384/iter
    const unsigned char* p_wr = &Pl[0][0] + (4 * wv) * P_STRIDE + 8 * lane;
    const unsigned char* p_rd = &Pl[0][0] + ln * P_STRIDE + 16 * lg;

    v8f acc = {};
    const int nt = GAT_N / 128;   // 64

    for (int t = 0; t < nt; ++t) {
        const float4 w2 = *(const float4*)w2ptr;
        const int    db = (t & 1) * P_BUF;

        // prefetch adj two tiles ahead (streamed once; cover L2/HBM latency)
        if (t + 2 < nt) {
#pragma unroll
            for (int r = 0; r < 4; ++r)
                __builtin_prefetch(aptr + r * GAT_N + 256, 0, 2);
        }

        // ---- score stage: lane = 4 columns (coalesced b128 adj), 4 rows per wave ----
#pragma unroll
        for (int r = 0; r < 4; ++r) {
            const int4 av = *(const int4*)(aptr + r * GAT_N);   // immediate offset r*32768B
            const float nM = nMl[r];
            const float w1 = wh1r[r];
            // 6 VALU/elem: add, mul, max, fma, cndmask, exp2 (unconditional -> branchless)
            float a0 = fmaf(leaky(w1 + w2.x), LOG2E, nM);  a0 = (av.x > 0) ? a0 : EXP2_NEG;
            float a1 = fmaf(leaky(w1 + w2.y), LOG2E, nM);  a1 = (av.y > 0) ? a1 : EXP2_NEG;
            float a2 = fmaf(leaky(w1 + w2.z), LOG2E, nM);  a2 = (av.z > 0) ? a2 : EXP2_NEG;
            float a3 = fmaf(leaky(w1 + w2.w), LOG2E, nM);  a3 = (av.w > 0) ? a3 : EXP2_NEG;
            const float p0 = __builtin_amdgcn_exp2f(a0);
            const float p1 = __builtin_amdgcn_exp2f(a1);
            const float p2 = __builtin_amdgcn_exp2f(a2);
            const float p3 = __builtin_amdgcn_exp2f(a3);
            lr[r] += (p0 + p1) + (p2 + p3);

            const uint32_t lo = f2bf_bits(p0) | (f2bf_bits(p1) << 16);
            const uint32_t hi = f2bf_bits(p2) | (f2bf_bits(p3) << 16);
            *(uint2*)(p_wr + db + r * P_STRIDE) = make_uint2(lo, hi);   // ds_store_b64
        }
        __syncthreads();   // single barrier/iter: double buffer protects buffer reuse

        // ---- 4 chained WMMAs over the 128-wide tile ----
#pragma unroll
        for (int s = 0; s < 4; ++s) {
            union { v16bf v; uint4 q[2]; } A;
            A.q[0] = *(const uint4*)(p_rd + db + 64 * s);        // K = 8g .. 8g+7
            A.q[1] = *(const uint4*)(p_rd + db + 64 * s + 32);   // K = 16+8g .. 23+8g

            union { v16bf v; uint4 q[2]; } B;
            B.q[0] = *(const uint4*)(bptr + s * 4096);
            B.q[1] = *(const uint4*)(bptr + s * 4096 + 16);

            acc = __builtin_amdgcn_wmma_f32_16x16x32_bf16(
                    false, A.v, false, B.v, (short)0, acc, false, false);
        }

        aptr  += 128;
        w2ptr += 128;
        bptr  += 16384;
    }

    // ---- one-time row-sum reduction across lanes ----
#pragma unroll
    for (int r = 0; r < 4; ++r) {
#pragma unroll
        for (int off = 16; off > 0; off >>= 1)
            lr[r] += __shfl_xor(lr[r], off, 32);
    }
    if (lane == 0) {
#pragma unroll
        for (int r = 0; r < 4; ++r) Ll[4 * wv + r] = lr[r];
    }
    __syncthreads();

    // ---- epilogue: divide by denom, ReLU, store per C/D f32 layout ----
#pragma unroll
    for (int k = 0; k < 8; ++k) {
        const int row = k + 8 * lg;
        const float denom = Ll[row];
        const float inv = (denom > 0.f) ? (1.f / denom) : 0.f;
        float val = acc[k] * inv;
        val = fmaxf(val, 0.f);
        out[(size_t)(i0 + row) * GAT_D + 16 * wv + ln] = val;
    }
}

// ---------------- Host-side launcher ----------------
extern "C" void kernel_launch(void* const* d_in, const int* in_sizes, int n_in,
                              void* d_out, int out_size, void* d_ws, size_t ws_size,
                              hipStream_t stream) {
    const float* h   = (const float*)d_in[0];
    const int*   adj = (const int*)d_in[1];
    const float* a   = (const float*)d_in[2];
    float* out = (float*)d_out;

    char* ws = (char*)d_ws;
    float*    wh1   = (float*)ws;                                       // N floats
    float*    wh2   = (float*)(ws + (size_t)GAT_N * 4);                 // N floats
    float*    w2max = (float*)(ws + (size_t)2 * GAT_N * 4);             // 1 float (+pad)
    uint32_t* bfrag = (uint32_t*)(ws + (size_t)2 * GAT_N * 4 + 16);     // N*32 dwords (1 MB)

    gat_wh_kernel<<<(GAT_N + 255) / 256, 256, 0, stream>>>(h, a, wh1, wh2);
    gat_w2max_kernel<<<1, 256, 0, stream>>>(wh2, w2max);

    const int nbf = (GAT_N / 32) * 4 * 32;
    gat_bfrag_kernel<<<(nbf + 255) / 256, 256, 0, stream>>>(h, bfrag);

    gat_main_kernel<<<GAT_N / 16, 128, 0, stream>>>(wh1, wh2, w2max, adj, bfrag, out);
}
